// MHAttention_21217138442394
// MI455X (gfx1250) — matrix operand order repaired and checked by reading, not compiled, not run on hardware
//
#include <hip/hip_runtime.h>
#include <hip/hip_bf16.h>

// MHA forward for MI455X (gfx1250): bf16 WMMA everywhere, fp32 accumulate.
// Stages: (1) f32->bf16 converts, (2) Q/K/V projections via WMMA GEMM
// (V stored transposed per head), (3) causal flash-attention with
// transposed-score trick (lane-local softmax), (4) output projection.
// GEMM wave tile is 64x64 (16 WMMAs per 8 fragment loads per k-step) to
// maximize matrix-pipe feed from WGP$/L2 (entire working set is L2-resident).

typedef __attribute__((ext_vector_type(16))) __bf16 v16bf;
typedef __attribute__((ext_vector_type(8)))  float  v8f;

#define TT 2048   // sequence length
#define HH 16     // heads
#define DH 128    // head dim
#define DD 2048   // hidden
#define KK 2048   // GEMM contraction

union Frag { uint4 u[2]; v16bf v; };

static __device__ inline v8f zero_v8f() {
  v8f z = {0.f,0.f,0.f,0.f,0.f,0.f,0.f,0.f};
  return z;
}

// A-operand fragment (16x32 bf16): lane l -> row (l&15); k-chunks at
// half*8 .. +7 and half*8+16 .. +23 (ISA 7.12.2, 16-bit A layout).
static __device__ inline v16bf load_a_frag(const __bf16* __restrict__ base,
                                           int row, int ldk, int kbase, int lane) {
  const int half = lane >> 4;
  const __bf16* p = base + (size_t)(row + (lane & 15)) * ldk + kbase + half * 8;
  Frag f;
  f.u[0] = *(const uint4*)(p);
  f.u[1] = *(const uint4*)(p + 16);
  return f.v;
}

// B-operand fragment (32x16 bf16): lane l -> column (l&15); k-chunks at
// half*8 .. +7 and half*8+16 .. +23 (same order as the A fragment).
static __device__ inline v16bf load_b_frag(const __bf16* __restrict__ base,
                                           int col, int ldk, int kbase, int lane) {
  const __bf16* p = base + (size_t)(col + (lane & 15)) * ldk + kbase + ((lane >> 4) << 3);
  Frag f;
  f.u[0] = *(const uint4*)(p);
  f.u[1] = *(const uint4*)(p + 16);
  return f.v;
}

static __device__ inline v8f wmma_bf16(v16bf a, v16bf b, v8f c) {
  return __builtin_amdgcn_wmma_f32_16x16x32_bf16(
      /*neg_a=*/false, a, /*neg_b=*/false, b,
      /*c_mod=*/(short)0, c, /*reuse_a=*/false, /*reuse_b=*/false);
}

// ---------------------------------------------------------------------------
// f32 -> bf16 conversion
// ---------------------------------------------------------------------------
__global__ void cvt_f32_to_bf16(const float* __restrict__ src,
                                __bf16* __restrict__ dst, int n) {
  int i = (blockIdx.x * blockDim.x + threadIdx.x) * 4;
  if (i + 3 < n) {
    float4 f = *(const float4*)(src + i);
    dst[i+0] = (__bf16)f.x;
    dst[i+1] = (__bf16)f.y;
    dst[i+2] = (__bf16)f.z;
    dst[i+3] = (__bf16)f.w;
  } else {
    for (; i < n; i++) dst[i] = (__bf16)src[i];
  }
}

// ---------------------------------------------------------------------------
// NT GEMM: C[m,n] = (sum_k A[m,k]*W[n,k] + bias[n]) * scale
// 256 threads = 8 waves (2 m-rows x 4 n-cols); block tile 128x256;
// wave tile 64x64 -> 16 accumulators, 16 WMMAs per 32-deep k-step.
// mode 0: bf16 row-major store; mode 1: bf16 per-head-transposed store
// (for V: [B,H,Dh,T]); mode 2: f32 row-major store (final output).
// ---------------------------------------------------------------------------
__global__ __launch_bounds__(256) void gemm_nt_bf16(
    const __bf16* __restrict__ A, const __bf16* __restrict__ W,
    const float* __restrict__ bias, void* __restrict__ out,
    int M, int N, int mode, float scale) {
  const int lane = threadIdx.x & 31;
  const int wave = threadIdx.x >> 5;
  const int m0 = blockIdx.y * 128 + (wave >> 2) * 64;
  const int n0 = blockIdx.x * 256 + (wave & 3) * 64;

  v8f acc[4][4];
  #pragma unroll
  for (int i = 0; i < 4; i++)
    #pragma unroll
    for (int j = 0; j < 4; j++) acc[i][j] = zero_v8f();

  for (int k = 0; k < KK; k += 32) {
    v16bf bf[4];
    #pragma unroll
    for (int j = 0; j < 4; j++) bf[j] = load_b_frag(W, n0 + j * 16, KK, k, lane);
    #pragma unroll
    for (int i = 0; i < 4; i++) {
      v16bf a = load_a_frag(A, m0 + i * 16, KK, k, lane);
      #pragma unroll
      for (int j = 0; j < 4; j++) acc[i][j] = wmma_bf16(a, bf[j], acc[i][j]);
    }
  }

  const int ln = lane & 15;
  const int ra = (lane >> 4) << 3;   // +8 row offset for upper half-wave
  #pragma unroll
  for (int i = 0; i < 4; i++) {
    #pragma unroll
    for (int j = 0; j < 4; j++) {
      int n = n0 + j * 16 + ln;
      float bv = bias[n];
      #pragma unroll
      for (int v = 0; v < 8; v++) {
        int m = m0 + i * 16 + ra + v;
        float val = (acc[i][j][v] + bv) * scale;
        if (mode == 0) {
          ((__bf16*)out)[(size_t)m * N + n] = (__bf16)val;
        } else if (mode == 1) {
          int bidx = m >> 11, t = m & (TT - 1);
          int h = n >> 7,     d = n & (DH - 1);
          ((__bf16*)out)[(((size_t)bidx * HH + h) * DH + d) * TT + t] = (__bf16)val;
        } else {
          ((float*)out)[(size_t)m * N + n] = val;
        }
      }
    }
  }
}

// ---------------------------------------------------------------------------
// Causal flash attention. 128 threads = 4 independent waves; each wave owns
// 16 query rows. Scores computed transposed (S' = K*Q^T) so softmax is
// lane-local and P repacks into the WMMA A-layout without cross-lane moves.
// Q was pre-scaled by 1/sqrt(DH) in its projection epilogue.
// ---------------------------------------------------------------------------
__global__ __launch_bounds__(128) void flash_attn(
    const __bf16* __restrict__ Qm, const __bf16* __restrict__ Km,
    const __bf16* __restrict__ Vt, __bf16* __restrict__ ctx) {
  const int lane = threadIdx.x & 31;
  const int wave = threadIdx.x >> 5;
  const int h = blockIdx.y;
  const int b = blockIdx.z;
  const int qbase = (blockIdx.x * 4 + wave) * 16;

  const __bf16* Qb = Qm + (size_t)b * TT * DD + h * DH;
  const __bf16* Kb = Km + (size_t)b * TT * DD + h * DH;
  const __bf16* Vb = Vt + ((size_t)b * HH + h) * DH * TT;

  v16bf qf[4];
  #pragma unroll
  for (int d = 0; d < 4; d++) qf[d] = load_b_frag(Qb, qbase, DD, d * 32, lane);

  v8f o[8];
  #pragma unroll
  for (int f = 0; f < 8; f++) o[f] = zero_v8f();
  float m_run = -__builtin_inff();
  float l_run = 0.f;

  const int qi = qbase + (lane & 15);  // this lane's query row
  const int ra = (lane >> 4) << 3;     // +8 offset for upper half-wave
  const int nblk = (qbase + 16 + 31) >> 5;  // 32-key blocks (causal bound)

  for (int kb = 0; kb < nblk; kb++) {
    const int kk0 = kb << 5;
    v8f s0 = zero_v8f();
    v8f s1 = zero_v8f();
    #pragma unroll
    for (int d = 0; d < 4; d++) {
      v16bf kf0 = load_a_frag(Kb, kk0,      DD, d * 32, lane);
      v16bf kf1 = load_a_frag(Kb, kk0 + 16, DD, d * 32, lane);
      s0 = wmma_bf16(kf0, qf[d], s0);   // S'[kk 0..15 , q]
      s1 = wmma_bf16(kf1, qf[d], s1);   // S'[kk 16..31, q]
    }
    if (kb == nblk - 1) {               // diagonal block: causal mask
      int kl = kk0 + ra;
      #pragma unroll
      for (int v = 0; v < 8; v++) {
        if (kl + v      > qi) s0[v] = -3.0e38f;
        if (kl + 16 + v > qi) s1[v] = -3.0e38f;
      }
    }
    // lane-local row max over this block's 16 scores (+ partner half)
    float mt = s0[0];
    #pragma unroll
    for (int v = 1; v < 8; v++) mt = fmaxf(mt, s0[v]);
    #pragma unroll
    for (int v = 0; v < 8; v++) mt = fmaxf(mt, s1[v]);
    mt = fmaxf(mt, __shfl_xor(mt, 16, 32));
    float m_new = fmaxf(m_run, mt);
    float alpha = __expf(m_run - m_new);
    m_run = m_new;

    // P = exp(S'-m), packed directly into the WMMA A-layout
    v16bf pa;
    float sum = 0.f;
    #pragma unroll
    for (int v = 0; v < 8; v++) {
      float p0 = __expf(s0[v] - m_new);
      float p1 = __expf(s1[v] - m_new);
      sum += p0 + p1;
      pa[v]     = (__bf16)p0;
      pa[v + 8] = (__bf16)p1;
    }
    sum += __shfl_xor(sum, 16, 32);
    l_run = l_run * alpha + sum;

    // rescale O accumulators: alpha of O-row (ra+v) lives in lane (ra+v)
    float al[8];
    #pragma unroll
    for (int v = 0; v < 8; v++) al[v] = __shfl(alpha, ra + v, 32);
    #pragma unroll
    for (int f = 0; f < 8; f++)
      #pragma unroll
      for (int v = 0; v < 8; v++) o[f][v] *= al[v];

    // O += P * V  (V pre-transposed: contiguous B fragments)
    #pragma unroll
    for (int f = 0; f < 8; f++) {
      v16bf vf = load_b_frag(Vb, f * 16, TT, kk0, lane);
      o[f] = wmma_bf16(pa, vf, o[f]);
    }
  }

  float linv = 1.f / l_run;
  float li[8];
  #pragma unroll
  for (int v = 0; v < 8; v++) li[v] = __shfl(linv, ra + v, 32);

  __bf16* Cb = ctx + (size_t)b * TT * DD + h * DH;
  #pragma unroll
  for (int f = 0; f < 8; f++) {
    int dcol = f * 16 + (lane & 15);
    #pragma unroll
    for (int v = 0; v < 8; v++) {
      int q = qbase + ra + v;
      Cb[(size_t)q * DD + dcol] = (__bf16)(o[f][v] * li[v]);
    }
  }
}

// ---------------------------------------------------------------------------
extern "C" void kernel_launch(void* const* d_in, const int* in_sizes, int n_in,
                              void* d_out, int out_size, void* d_ws, size_t ws_size,
                              hipStream_t stream) {
  const float* x  = (const float*)d_in[0];
  const float* Wq = (const float*)d_in[1];
  const float* bq = (const float*)d_in[2];
  const float* Wk = (const float*)d_in[3];
  const float* bk = (const float*)d_in[4];
  const float* Wv = (const float*)d_in[5];
  const float* bv = (const float*)d_in[6];
  const float* Wo = (const float*)d_in[7];
  const float* bo = (const float*)d_in[8];

  const int M = 2 * TT;  // B*T = 4096 token rows

  char* p = (char*)d_ws;
  __bf16* xb   = (__bf16*)p; p += (size_t)M  * DD * 2;
  __bf16* wqb  = (__bf16*)p; p += (size_t)DD * DD * 2;
  __bf16* wkb  = (__bf16*)p; p += (size_t)DD * DD * 2;
  __bf16* wvb  = (__bf16*)p; p += (size_t)DD * DD * 2;
  __bf16* wob  = (__bf16*)p; p += (size_t)DD * DD * 2;
  __bf16* qb   = (__bf16*)p; p += (size_t)M  * DD * 2;
  __bf16* kbuf = (__bf16*)p; p += (size_t)M  * DD * 2;
  __bf16* vtb  = (__bf16*)p; p += (size_t)M  * DD * 2;   // [B,H,Dh,T]
  __bf16* ctxb = (__bf16*)p; p += (size_t)M  * DD * 2;

  const int threads = 256;
  {
    int n = M * DD;
    cvt_f32_to_bf16<<<(n / 4 + threads - 1) / threads, threads, 0, stream>>>(x, xb, n);
    n = DD * DD;
    int blk = (n / 4 + threads - 1) / threads;
    cvt_f32_to_bf16<<<blk, threads, 0, stream>>>(Wq, wqb, n);
    cvt_f32_to_bf16<<<blk, threads, 0, stream>>>(Wk, wkb, n);
    cvt_f32_to_bf16<<<blk, threads, 0, stream>>>(Wv, wvb, n);
    cvt_f32_to_bf16<<<blk, threads, 0, stream>>>(Wo, wob, n);
  }

  dim3 gg(DD / 256, M / 128, 1);
  dim3 bb(256);
  const float qscale = 0.08838834764831845f;  // 1/sqrt(128), folded into Q
  gemm_nt_bf16<<<gg, bb, 0, stream>>>(xb, wqb, bq, qb,   M, DD, 0, qscale);
  gemm_nt_bf16<<<gg, bb, 0, stream>>>(xb, wkb, bk, kbuf, M, DD, 0, 1.0f);
  gemm_nt_bf16<<<gg, bb, 0, stream>>>(xb, wvb, bv, vtb,  M, DD, 1, 1.0f);

  flash_attn<<<dim3(TT / 64, HH, 2), dim3(128), 0, stream>>>(qb, kbuf, vtb, ctxb);

  gemm_nt_bf16<<<gg, bb, 0, stream>>>(ctxb, wob, bo, d_out, M, DD, 2, 1.0f);
}
